// ManualLSTM_37280316129802
// MI455X (gfx1250) — matrix operand order repaired and checked by reading, not compiled, not run on hardware
//
#include <hip/hip_runtime.h>
#include <math.h>

// ---------------------------------------------------------------------------
// LSTM on MI455X (gfx1250): bf16 WMMA, fp32 accum/state.
// Phase 1: xg = x @ Wx^T  (M=65536,N=2048,K=512) -- tiled WMMA GEMM from LDS.
// Phase 2: 16 persistent WGs; TDM loads weight slice (128KB) resident into
//          LDS with hardware padding; TDM stages h (64KB) per step; c kept in
//          registers; global-atomic barrier per timestep; ping-pong h buffer.
// ---------------------------------------------------------------------------

typedef __attribute__((ext_vector_type(16))) __bf16 v16bf;
typedef __attribute__((ext_vector_type(8)))  __bf16 v8bf;
typedef __attribute__((ext_vector_type(8)))  float  v8f;
typedef __attribute__((ext_vector_type(4)))  unsigned int u32x4;
typedef __attribute__((ext_vector_type(8)))  int i32x8;
typedef __attribute__((ext_vector_type(4)))  int i32x4;

#define BB 64
#define TT 1024
#define DD 512
#define HH 512
#define G4 2048          // 4*H
#define MTOT (BB*TT)     // 65536

__device__ __forceinline__ float fast_sigmoid(float x) {
    return 1.0f / (1.0f + __expf(-x));
}
__device__ __forceinline__ float fast_tanh(float x) {
    float ax = fabsf(x);
    float e  = __expf(-2.0f * ax);
    float t  = (1.0f - e) / (1.0f + e);
    return copysignf(t, x);
}

// ---------------------------------------------------------------------------
// Tensor Data Mover: build D# and issue TENSOR_LOAD_TO_LDS (6-arg builtin).
//   - 2D/3D tile of bf16 (data_size=2B), row length tile_d0 elems,
//     tile_d1 rows, optional tile_d2 blocks stepping by stride1 elems.
//   - pad_enable: +4 DWORDs every 256 DWORDs  => LDS row stride 520 elems
//     for 512-elem rows (bank-conflict-free fragment reads).
// ---------------------------------------------------------------------------
__device__ __forceinline__ void tdm_load_bf16(
    unsigned lds_addr, const void* gaddr,
    unsigned tile_d0, unsigned tile_d1, unsigned tile_d2,
    unsigned long long stride0, unsigned long long stride1,
    unsigned tdim0, unsigned tdim1, unsigned tdim2)
{
    unsigned long long ga = (unsigned long long)gaddr;
    u32x4 g0;
    g0[0] = 1u;                                                 // count=1, user desc
    g0[1] = lds_addr;                                           // LDS byte address
    g0[2] = (unsigned)(ga & 0xFFFFFFFFu);                       // global_addr[31:0]
    g0[3] = (unsigned)((ga >> 32) & 0x1FFFFFFu) | (2u << 30);   // addr[56:32] | type=2
    i32x8 g1;
    g1[0] = (int)((1u << 16)      // data_size = 2 bytes
                | (1u << 20)      // pad_enable
                | (7u << 22)      // pad_interval: 256 DWORDs (=1024B row)
                | (3u << 25));    // pad_amount: 4 DWORDs (=16B -> stride 520)
    g1[1] = (int)((tdim0 & 0xFFFFu) << 16);                     // abar=0 | tdim0.lo
    g1[2] = (int)((tdim0 >> 16) | ((tdim1 & 0xFFFFu) << 16));
    g1[3] = (int)((tdim1 >> 16) | (tile_d0 << 16));
    g1[4] = (int)(tile_d1 | (tile_d2 << 16));
    g1[5] = (int)(stride0 & 0xFFFFFFFFu);
    g1[6] = (int)(((stride0 >> 32) & 0xFFFFu) | ((stride1 & 0xFFFFu) << 16));
    g1[7] = (int)((stride1 >> 16) & 0xFFFFFFFFu);
    i32x4 g2;
    g2[0] = (int)tdim2; g2[1] = 0; g2[2] = 0; g2[3] = 0;        // tensor_dim2
    i32x4 g3;
    g3[0] = 0; g3[1] = 0; g3[2] = 0; g3[3] = 0;
    i32x8 g4;                                                    // extra group (zero)
    g4[0] = 0; g4[1] = 0; g4[2] = 0; g4[3] = 0;
    g4[4] = 0; g4[5] = 0; g4[6] = 0; g4[7] = 0;
    __builtin_amdgcn_tensor_load_to_lds(g0, g1, g2, g3, g4, 0);
}

// ---------------------------------------------------------------------------
// Prep: x [B,T,D] f32  ->  xb [T*B, D] bf16   (m = t*B + b)
// ---------------------------------------------------------------------------
__global__ void prep_x_kernel(const float* __restrict__ x, __bf16* __restrict__ xb) {
    size_t i = ((size_t)blockIdx.x * blockDim.x + threadIdx.x) * 8;  // 8 elems/thread
    if (i >= (size_t)BB * TT * DD) return;
    int    d  = (int)(i % DD);
    size_t bt = i / DD;
    int    t  = (int)(bt % TT);
    int    b  = (int)(bt / TT);
    __bf16 tmp[8];
#pragma unroll
    for (int e = 0; e < 8; ++e) tmp[e] = (__bf16)x[i + e];
    *(int4*)(xb + ((size_t)t * BB + b) * DD + d) = *(const int4*)tmp;
}

// ---------------------------------------------------------------------------
// Prep: weights -> bf16 concat [i,f,o,g]; bias concat f32; zero h ping-pong
// and the global barrier counter (re-run every launch -> deterministic).
// ---------------------------------------------------------------------------
__global__ void prep_w_kernel(
    const float* __restrict__ Wii, const float* __restrict__ Whi, const float* __restrict__ bi,
    const float* __restrict__ Wif, const float* __restrict__ Whf, const float* __restrict__ bff,
    const float* __restrict__ Wio, const float* __restrict__ Who, const float* __restrict__ bo,
    const float* __restrict__ Wig, const float* __restrict__ Whg, const float* __restrict__ bg,
    __bf16* __restrict__ Wx, __bf16* __restrict__ Wh, float* __restrict__ bias,
    __bf16* __restrict__ ghbuf, unsigned* __restrict__ cnt)
{
    int i = blockIdx.x * blockDim.x + threadIdx.x;
    const int WN = G4 * DD;  // 2048*512
    if (i < WN) {
        int g = i / (HH * DD);
        int r = i % (HH * DD);
        const float* sx = (g == 0) ? Wii : (g == 1) ? Wif : (g == 2) ? Wio : Wig;
        const float* sh = (g == 0) ? Whi : (g == 1) ? Whf : (g == 2) ? Who : Whg;
        Wx[i] = (__bf16)sx[r];
        Wh[i] = (__bf16)sh[r];
    }
    if (i < G4) {
        int g = i >> 9, r = i & 511;
        const float* sb = (g == 0) ? bi : (g == 1) ? bff : (g == 2) ? bo : bg;
        bias[i] = sb[r];
    }
    if (i < 2 * BB * HH) ghbuf[i] = (__bf16)0.0f;   // zero both ping-pong h buffers
    if (i == 0) *cnt = 0u;
}

// ---------------------------------------------------------------------------
// Phase 1 GEMM: xg[m][n] = sum_k xb[m][k] * Wx[n][k]   (fp32 out)
// Block tile 128x128, K-tile 64, 8 waves (4x2), wave tile 32x64.
// ---------------------------------------------------------------------------
#define LDT 72   // padded LDS row stride (elems) -> conflict-free frag reads

__global__ __launch_bounds__(256) void xproj_gemm_kernel(
    const __bf16* __restrict__ A,   // [MTOT, 512]
    const __bf16* __restrict__ Wx,  // [2048, 512]
    float* __restrict__ xg)         // [MTOT, 2048]
{
    __shared__ __bf16 As[128 * LDT];
    __shared__ __bf16 Bs[128 * LDT];

    const int m0   = blockIdx.x * 128;
    const int n0   = blockIdx.y * 128;
    const int tid  = threadIdx.x;
    const int wave = tid >> 5, lane = tid & 31;
    const int wm   = wave & 3, wn = wave >> 2;     // 4 x 2 wave grid
    const int ln   = lane & 15, lh = lane >> 4;

    v8f acc[2][4] = {};

    for (int k0 = 0; k0 < DD; k0 += 64) {
        __syncthreads();
        // cooperative stage: 128 rows x 64 cols bf16 for both A and B slices
        for (int c = tid; c < 128 * 8; c += 256) {           // 16B chunks
            int r = c >> 3, off = (c & 7) * 8;
            *(int4*)(As + r * LDT + off) = *(const int4*)(A  + (size_t)(m0 + r) * DD + k0 + off);
            *(int4*)(Bs + r * LDT + off) = *(const int4*)(Wx + (size_t)(n0 + r) * DD + k0 + off);
        }
        __syncthreads();

#pragma unroll
        for (int ks = 0; ks < 64; ks += 32) {
            v16bf a[2], b[4];
#pragma unroll
            for (int i2 = 0; i2 < 2; ++i2) {
                const __bf16* ar = As + (wm * 32 + i2 * 16 + ln) * LDT + ks + lh * 8;
                v8bf lo = *(const v8bf*)ar;
                v8bf hi = *(const v8bf*)(ar + 16);
#pragma unroll
                for (int e = 0; e < 8; ++e) { a[i2][e] = lo[e]; a[i2][8 + e] = hi[e]; }
            }
#pragma unroll
            for (int j2 = 0; j2 < 4; ++j2) {
                const __bf16* br = Bs + (wn * 64 + j2 * 16 + ln) * LDT + ks + lh * 8;
                v8bf lo = *(const v8bf*)br;
                v8bf hi = *(const v8bf*)(br + 16);
#pragma unroll
                for (int e = 0; e < 8; ++e) { b[j2][e] = lo[e]; b[j2][8 + e] = hi[e]; }
            }
#pragma unroll
            for (int i2 = 0; i2 < 2; ++i2)
#pragma unroll
                for (int j2 = 0; j2 < 4; ++j2)
                    acc[i2][j2] = __builtin_amdgcn_wmma_f32_16x16x32_bf16(
                        false, a[i2], false, b[j2], (short)0, acc[i2][j2], false, false);
        }
    }

#pragma unroll
    for (int i2 = 0; i2 < 2; ++i2)
#pragma unroll
        for (int j2 = 0; j2 < 4; ++j2)
#pragma unroll
            for (int e = 0; e < 8; ++e) {
                int m = m0 + wm * 32 + i2 * 16 + e + 8 * lh;
                int n = n0 + wn * 64 + j2 * 16 + ln;
                xg[(size_t)m * G4 + n] = acc[i2][j2][e];
            }
}

// ---------------------------------------------------------------------------
// Phase 2: persistent recurrence. 16 WGs, each owns 32 columns of H.
// TDM pulls the weight slice [4 x 32 x 512] bf16 into LDS once (3D tile with
// hardware row padding -> stride 520), and stages h [64 x 512] each step.
// ---------------------------------------------------------------------------
#define WSTR 520   // LDS row stride (elems) produced by TDM padding
#define HSTR 520
#define REC_LDS_BYTES (128 * WSTR * 2 + 64 * HSTR * 2)   // 199,680 B < 320KB

__global__ __launch_bounds__(256, 1) void lstm_rec_kernel(
    const float*  __restrict__ xg,     // [T*B, 2048]
    const __bf16* __restrict__ Wh,     // [2048, 512] bf16
    const float*  __restrict__ bias,   // [2048]
    __bf16*       __restrict__ ghbuf,  // [2][B*H] ping-pong
    unsigned*     __restrict__ cnt,
    float*        __restrict__ out)    // [B,T,H] ++ h_T [B,H] ++ c_T [B,H]
{
    extern __shared__ char smem[];
    __bf16* Wl = (__bf16*)smem;                        // 128 x WSTR
    __bf16* hl = (__bf16*)(smem + 128 * WSTR * 2);     // 64  x HSTR
    const unsigned lds_base = __builtin_amdgcn_groupstaticsize();  // dyn-LDS offset
    const unsigned wl_off = lds_base;
    const unsigned hl_off = lds_base + 128 * WSTR * 2;

    const int wg    = blockIdx.x;          // 0..15 -> H chunk
    const int j0    = wg * 32;
    const int tid   = threadIdx.x;
    const int wave  = tid >> 5, lane = tid & 31;
    const int mtile = wave & 3, jhalf = wave >> 2;   // 4 m-tiles x 2 j-halves
    const int m0    = mtile * 16;
    const int ln    = lane & 15, lh = lane >> 4;
    const int jcol  = j0 + jhalf * 16 + ln;          // this lane's h column

    // ---- TDM: resident weight slice, rows {g*512 + j0 .. +31}, g=0..3 ------
    // 3D tile: x=512, y=32 rows (stride 512), z=4 gate blocks (stride 512*512)
    if (tid < 32) {
        tdm_load_bf16(wl_off, Wh + (size_t)j0 * DD,
                      /*tile*/ 512u, 32u, 4u,
                      /*strides*/ 512ull, 512ull * 512ull,
                      /*tensor dims*/ 512u, 2048u, 4u);
    }
    __builtin_amdgcn_s_wait_tensorcnt(0);

    float bvals[4];
#pragma unroll
    for (int g = 0; g < 4; ++g) bvals[g] = bias[g * HH + jcol];

    float c_reg[8], h_reg[8];
#pragma unroll
    for (int e = 0; e < 8; ++e) { c_reg[e] = 0.0f; h_reg[e] = 0.0f; }

    int p = 0;  // read buffer parity
    for (int t = 0; t < TT; ++t) {
        // ---- device-wide barrier: h writes of previous step now visible ----
        __threadfence();
        __syncthreads();
        if (tid == 0) {
            __hip_atomic_fetch_add(cnt, 1u, __ATOMIC_ACQ_REL, __HIP_MEMORY_SCOPE_AGENT);
            unsigned want = (unsigned)(t + 1) * 16u;
            while (__hip_atomic_load(cnt, __ATOMIC_ACQUIRE, __HIP_MEMORY_SCOPE_AGENT) < want)
                __builtin_amdgcn_s_sleep(2);
        }
        __syncthreads();

        // ---- TDM: stage h_t (full [64,512] bf16) into LDS ------------------
        if (tid < 32) {
            const __bf16* hsrc = ghbuf + (size_t)p * (BB * HH);
            tdm_load_bf16(hl_off, hsrc,
                          /*tile*/ 512u, 64u, 0u,
                          /*strides*/ 512ull, 0ull,
                          /*tensor dims*/ 512u, 64u, 0u);
        }
        __builtin_amdgcn_s_wait_tensorcnt(0);
        __syncthreads();

        // ---- gates_chunk += h @ Wh_slice^T : 4 accumulators (one per gate) --
        v8f acc[4] = {};
        for (int kk = 0; kk < DD; kk += 32) {
            v16bf a;
            {
                const __bf16* ar = hl + (m0 + ln) * HSTR + kk + lh * 8;
                v8bf lo = *(const v8bf*)ar;
                v8bf hi = *(const v8bf*)(ar + 16);
#pragma unroll
                for (int e = 0; e < 8; ++e) { a[e] = lo[e]; a[8 + e] = hi[e]; }
            }
#pragma unroll
            for (int g = 0; g < 4; ++g) {
                const __bf16* br = Wl + (g * 32 + jhalf * 16 + ln) * WSTR + kk + lh * 8;
                v8bf lo = *(const v8bf*)br;
                v8bf hi = *(const v8bf*)(br + 16);
                v16bf b;
#pragma unroll
                for (int e = 0; e < 8; ++e) { b[e] = lo[e]; b[8 + e] = hi[e]; }
                acc[g] = __builtin_amdgcn_wmma_f32_16x16x32_bf16(
                    false, a, false, b, (short)0, acc[g], false, false);
            }
        }

        // ---- nonlinearities + state update; emit h_{t+1} -------------------
        __bf16* hdst = ghbuf + (size_t)(1 - p) * (BB * HH);
#pragma unroll
        for (int e = 0; e < 8; ++e) {
            int brow = m0 + e + 8 * lh;                       // batch row
            const float* xgp = xg + ((size_t)t * BB + brow) * G4;
            float gi = acc[0][e] + xgp[0 * HH + jcol] + bvals[0];
            float gf = acc[1][e] + xgp[1 * HH + jcol] + bvals[1];
            float go = acc[2][e] + xgp[2 * HH + jcol] + bvals[2];
            float gg = acc[3][e] + xgp[3 * HH + jcol] + bvals[3];
            float i_s = fast_sigmoid(gi);
            float f_s = fast_sigmoid(gf);
            float o_s = fast_sigmoid(go);
            float g_t = fast_tanh(gg);
            float cn  = f_s * c_reg[e] + i_s * g_t;
            float hn  = o_s * fast_tanh(cn);
            c_reg[e]  = cn;
            h_reg[e]  = hn;
            out[((size_t)brow * TT + t) * HH + jcol] = hn;    // hs[b][t][j]
            hdst[brow * HH + jcol] = (__bf16)hn;              // h for next step
            // warm L2->WGP path for next step's gate pre-activations
            if (t + 1 < TT)
                __builtin_prefetch(xgp + (size_t)BB * G4, 0, 1);
        }
        p ^= 1;
    }

    // ---- final (h_T, c_T) appended after hs --------------------------------
    float* out_h = out + (size_t)BB * TT * HH;
    float* out_c = out_h + (size_t)BB * HH;
#pragma unroll
    for (int e = 0; e < 8; ++e) {
        int brow = m0 + e + 8 * lh;
        out_h[brow * HH + jcol] = h_reg[e];
        out_c[brow * HH + jcol] = c_reg[e];
    }
}

// ---------------------------------------------------------------------------
extern "C" void kernel_launch(void* const* d_in, const int* in_sizes, int n_in,
                              void* d_out, int out_size, void* d_ws, size_t ws_size,
                              hipStream_t stream) {
    const float* x   = (const float*)d_in[0];
    const float* Wii = (const float*)d_in[1];
    const float* Whi = (const float*)d_in[2];
    const float* bi  = (const float*)d_in[3];
    const float* Wif = (const float*)d_in[4];
    const float* Whf = (const float*)d_in[5];
    const float* bff = (const float*)d_in[6];
    const float* Wio = (const float*)d_in[7];
    const float* Who = (const float*)d_in[8];
    const float* bo  = (const float*)d_in[9];
    const float* Wig = (const float*)d_in[10];
    const float* Whg = (const float*)d_in[11];
    const float* bg  = (const float*)d_in[12];
    float* out = (float*)d_out;

    // workspace carve-up
    char* ws = (char*)d_ws;
    size_t off = 0;
    float*  xg   = (float*)(ws + off);  off += (size_t)MTOT * G4 * sizeof(float);   // 512 MB
    __bf16* xb   = (__bf16*)(ws + off); off += (size_t)MTOT * DD * 2;               // 64 MB
    __bf16* Wx   = (__bf16*)(ws + off); off += (size_t)G4 * DD * 2;                 // 2 MB
    __bf16* Wh   = (__bf16*)(ws + off); off += (size_t)G4 * DD * 2;                 // 2 MB
    float*  bias = (float*)(ws + off);  off += (size_t)G4 * sizeof(float);
    __bf16* ghb  = (__bf16*)(ws + off); off += (size_t)2 * BB * HH * 2;
    unsigned* cnt = (unsigned*)(ws + off);

    prep_w_kernel<<<(G4 * DD + 255) / 256, 256, 0, stream>>>(
        Wii, Whi, bi, Wif, Whf, bff, Wio, Who, bo, Wig, Whg, bg,
        Wx, Wh, bias, ghb, cnt);

    prep_x_kernel<<<((size_t)BB * TT * DD / 8 + 255) / 256, 256, 0, stream>>>(x, xb);

    dim3 ggrid(MTOT / 128, G4 / 128);   // (512, 16)
    xproj_gemm_kernel<<<ggrid, 256, 0, stream>>>(xb, Wx, xg);

    (void)hipFuncSetAttribute((const void*)lstm_rec_kernel,
                              hipFuncAttributeMaxDynamicSharedMemorySize,
                              REC_LDS_BYTES);
    lstm_rec_kernel<<<16, 256, REC_LDS_BYTES, stream>>>(xg, Wh, bias, ghb, cnt, out);
}
